// MyModel_87522843558773
// MI455X (gfx1250) — matrix operand order repaired and checked by reading, not compiled, not run on hardware
//
#include <hip/hip_runtime.h>

typedef __attribute__((ext_vector_type(16))) _Float16 v16h;
typedef __attribute__((ext_vector_type(8)))  float    v8f;

union Frag16 { v16h v; uint4 q[2]; };

#define UNITS   50
#define FEAT    128
#define NPAD    256      // 4*UNITS = 200 padded to 16 tiles of 16 (2 per wave, uniform)
#define KH      64       // UNITS padded to WMMA K granularity
#define T_STEPS 1024
#define MTILE   16
#define THREADS 256      // 8 waves (wave32)

// Padded LDS row strides (f16 elems / f32 elems) -> bank-conflict-free b128 loads:
// byte strides 272 (=17 granules of 16B) and 144 (=9 granules) are coprime with the
// 16-granule bank period; 260 f32 = 1040 B == 16 mod 256 for lZ.
#define XS  (FEAT + 8)   // 136 f16, 272 B
#define RS  (KH + 8)     // 72 f16, 144 B
#define ZS  260          // f32

__device__ __forceinline__ float fast_sigmoid(float x) {
    return 1.f / (1.f + __expf(-x));
}
__device__ __forceinline__ float fast_tanh(float x) {
    return 2.f / (1.f + __expf(-2.f * x)) - 1.f;
}

// Workgroup barrier with LDS-only ordering: does NOT drain LOADcnt, so the
// in-flight x_{t+1} global loads sail across it (unlike __syncthreads()).
__device__ __forceinline__ void barrier_lds_only() {
    asm volatile("s_wait_dscnt 0x0\n\t"
                 "s_barrier_signal -1\n\t"
                 "s_barrier_wait -1" ::: "memory");
}

__global__ __launch_bounds__(THREADS)
void lstm_persistent(const float* __restrict__ x,     // [512,1024,128]
                     const float* __restrict__ kern,  // [128,200]
                     const float* __restrict__ rker,  // [50,200]
                     const float* __restrict__ bias,  // [200]
                     const float* __restrict__ dw,    // [50]
                     const float* __restrict__ db,    // [1]
                     float* __restrict__ out)         // [512]
{
    __shared__ _Float16 lKT[NPAD * XS];       // 69632 B  kernel^T, f16, zero-padded
    __shared__ _Float16 lRT[NPAD * RS];       // 36864 B  rec_kernel^T, f16, zero-padded
    __shared__ _Float16 lX [2][MTILE * XS];   //  8704 B  double-buffered x_t tile
    __shared__ _Float16 lH [MTILE * RS];      //  2304 B  h state (f16)
    __shared__ float    lC [MTILE * KH];      //  4096 B  c state (f32)
    __shared__ float    lHF[MTILE * KH];      //  4096 B  h state (f32, dense head)
    __shared__ float    lZ [MTILE * ZS];      // 16640 B  pre-activation gates

    const int tid = threadIdx.x;
    const int b0  = blockIdx.x * MTILE;       // batch rows [b0, b0+16)

    // ---- one-time: weights -> LDS (f16, transposed [N][K], zero padded) ----
    for (int i = tid; i < NPAD * XS; i += THREADS) {
        int n = i / XS, k = i - n * XS;
        lKT[i] = (_Float16)((n < 4 * UNITS && k < FEAT) ? kern[k * (4 * UNITS) + n] : 0.f);
    }
    for (int i = tid; i < NPAD * RS; i += THREADS) {
        int n = i / RS, k = i - n * RS;
        lRT[i] = (_Float16)((n < 4 * UNITS && k < UNITS) ? rker[k * (4 * UNITS) + n] : 0.f);
    }
    for (int i = tid; i < MTILE * RS; i += THREADS) lH[i] = (_Float16)0.f;
    for (int i = tid; i < MTILE * KH; i += THREADS) { lC[i] = 0.f; lHF[i] = 0.f; }

    const int lane = tid & 31;
    const int wave = tid >> 5;     // 0..7: owns N-tiles wave and wave+8
    const int half = lane >> 4;    // ISA fragment half (lanes 0-15 vs 16-31)
    const int nlo  = lane & 15;    // M (A) / N (B,C,D) index within tile
    const int nb0  = wave * 16;
    const int nb1  = (wave + 8) * 16;

    float bias0, bias1;            // N column is constant per lane in C/D layout
    {
        int c0 = nb0 + nlo, c1 = nb1 + nlo;
        bias0 = (c0 < 4 * UNITS) ? bias[c0] : 0.f;
        bias1 = (c1 < 4 * UNITS) ? bias[c1] : 0.f;
    }

    // ---- register staging of the x stream: 2 float4 per thread, uniform ----
    const int i0 = tid, i1 = tid + THREADS;           // 512 float4 per step
    const int r0 = i0 >> 5, q0 = i0 & 31;
    const int r1 = i1 >> 5, q1 = i1 & 31;
    const float* p0 = x + ((size_t)(b0 + r0) * T_STEPS) * FEAT + q0 * 4;
    const float* p1 = x + ((size_t)(b0 + r1) * T_STEPS) * FEAT + q1 * 4;
    const int o0 = r0 * XS + q0 * 4;
    const int o1 = r1 * XS + q1 * 4;

    float4 v0 = *(const float4*)(p0);                 // preload t = 0
    float4 v1 = *(const float4*)(p1);

    __syncthreads();  // weights ready (full-fence barrier OK outside the loop)

    for (int t = 0; t < T_STEPS; ++t) {
        const int buf = t & 1;

        // ---- commit staged x_t (wait lands here), then launch x_{t+1} loads ----
        {
            _Float16* d0 = &lX[buf][o0];
            d0[0] = (_Float16)v0.x; d0[1] = (_Float16)v0.y;
            d0[2] = (_Float16)v0.z; d0[3] = (_Float16)v0.w;
            _Float16* d1 = &lX[buf][o1];
            d1[0] = (_Float16)v1.x; d1[1] = (_Float16)v1.y;
            d1[2] = (_Float16)v1.z; d1[3] = (_Float16)v1.w;
        }
        if (t + 1 < T_STEPS) {
            v0 = *(const float4*)(p0 + (size_t)(t + 1) * FEAT);
            v1 = *(const float4*)(p1 + (size_t)(t + 1) * FEAT);
            if (t + 4 < T_STEPS) {
                __builtin_prefetch(p0 + (size_t)(t + 4) * FEAT, 0, 0);
                __builtin_prefetch(p1 + (size_t)(t + 4) * FEAT, 0, 0);
            }
        }
        barrier_lds_only();   // lX[buf]/lH ready, lZ free; x_{t+1} loads stay in flight

        // ---- z = x@K + h@R + bias: two N-tiles per wave, interleaved accumulators ----
        {
            v8f acc0 = { bias0, bias0, bias0, bias0, bias0, bias0, bias0, bias0 };
            v8f acc1 = { bias1, bias1, bias1, bias1, bias1, bias1, bias1, bias1 };
            #pragma unroll
            for (int kf = 0; kf < 4; ++kf) {          // x @ kernel : K = 128
                const int kb = kf * 32;
                Frag16 a, bA, bB;
                a.q[0]  = *(const uint4*)&lX[buf][nlo * XS + kb + half * 16];
                a.q[1]  = *(const uint4*)&lX[buf][nlo * XS + kb + half * 16 + 8];
                bA.q[0] = *(const uint4*)&lKT[(nb0 + nlo) * XS + kb + half * 16];
                bA.q[1] = *(const uint4*)&lKT[(nb0 + nlo) * XS + kb + half * 16 + 8];
                bB.q[0] = *(const uint4*)&lKT[(nb1 + nlo) * XS + kb + half * 16];
                bB.q[1] = *(const uint4*)&lKT[(nb1 + nlo) * XS + kb + half * 16 + 8];
                acc0 = __builtin_amdgcn_wmma_f32_16x16x32_f16(false, a.v, false, bA.v,
                                                              (short)0, acc0, false, false);
                acc1 = __builtin_amdgcn_wmma_f32_16x16x32_f16(false, a.v, false, bB.v,
                                                              (short)0, acc1, false, false);
            }
            #pragma unroll
            for (int kf = 0; kf < 2; ++kf) {          // h @ rec_kernel : K = 64 (padded)
                const int kb = kf * 32;
                Frag16 a, bA, bB;
                a.q[0]  = *(const uint4*)&lH[nlo * RS + kb + half * 16];
                a.q[1]  = *(const uint4*)&lH[nlo * RS + kb + half * 16 + 8];
                bA.q[0] = *(const uint4*)&lRT[(nb0 + nlo) * RS + kb + half * 16];
                bA.q[1] = *(const uint4*)&lRT[(nb0 + nlo) * RS + kb + half * 16 + 8];
                bB.q[0] = *(const uint4*)&lRT[(nb1 + nlo) * RS + kb + half * 16];
                bB.q[1] = *(const uint4*)&lRT[(nb1 + nlo) * RS + kb + half * 16 + 8];
                acc0 = __builtin_amdgcn_wmma_f32_16x16x32_f16(false, a.v, false, bA.v,
                                                              (short)0, acc0, false, false);
                acc1 = __builtin_amdgcn_wmma_f32_16x16x32_f16(false, a.v, false, bB.v,
                                                              (short)0, acc1, false, false);
            }
            #pragma unroll
            for (int v = 0; v < 8; ++v) {             // C/D layout: N = nlo, M = v + 8*half
                lZ[(v + half * 8) * ZS + nb0 + nlo] = acc0[v];
                lZ[(v + half * 8) * ZS + nb1 + nlo] = acc1[v];
            }
        }
        barrier_lds_only();   // lZ complete

        // ---- gates + state update ----
        for (int e = tid; e < MTILE * UNITS; e += THREADS) {
            int m = e / UNITS, u = e - m * UNITS;
            float zi = lZ[m * ZS + u];
            float zf = lZ[m * ZS + UNITS + u];
            float zg = lZ[m * ZS + 2 * UNITS + u];
            float zo = lZ[m * ZS + 3 * UNITS + u];
            float c  = fast_sigmoid(zf) * lC[m * KH + u]
                     + fast_sigmoid(zi) * fast_tanh(zg);
            lC[m * KH + u] = c;
            float h  = fast_sigmoid(zo) * fast_tanh(c);
            lHF[m * KH + u] = h;
            lH [m * RS + u] = (_Float16)h;
        }
        // next iteration's first barrier orders lH writes vs WMMA reads
    }
    __syncthreads();

    // ---- dense head: out[b] = h_last . dense_w + dense_b ----
    if (tid < MTILE) {
        float s = db[0];
        #pragma unroll
        for (int u = 0; u < UNITS; ++u) s += lHF[tid * KH + u] * dw[u];
        out[b0 + tid] = s;
    }
}

extern "C" void kernel_launch(void* const* d_in, const int* in_sizes, int n_in,
                              void* d_out, int out_size, void* d_ws, size_t ws_size,
                              hipStream_t stream) {
    const float* x    = (const float*)d_in[0];
    const float* kern = (const float*)d_in[1];
    const float* rker = (const float*)d_in[2];
    const float* bias = (const float*)d_in[3];
    const float* dw   = (const float*)d_in[4];
    const float* db   = (const float*)d_in[5];
    float* out        = (float*)d_out;

    hipLaunchKernelGGL(lstm_persistent, dim3(512 / MTILE), dim3(THREADS), 0, stream,
                       x, kern, rker, bias, dw, db, out);
}
